// CLMCLLoss_10531259810062
// MI455X (gfx1250) — matrix operand order repaired and checked by reading, not compiled, not run on hardware
//
#include <hip/hip_runtime.h>

typedef __attribute__((ext_vector_type(16))) _Float16 v16h;
typedef __attribute__((ext_vector_type(8)))  _Float16 v8h;
typedef __attribute__((ext_vector_type(4)))  _Float16 v4h;
typedef __attribute__((ext_vector_type(8)))  float    v8f;

#define DIM      128
#define NROWS    1024
#define NCLS     50000
#define NTILES   3125          // 50000 / 16 column tiles
#define CCH      64            // C-chunks (grid.y); tile stride = 64
#define RB       4             // row-blocks (grid.x); 256 rows each
#define S_SCALE  30.0f
#define MARGIN   0.35f
#define LAMBDA   0.01f
// exp(30*c - 30) == exp2(fma(c, K_A, -K_A)),  K_A = 30*log2(e).
// Argument range [-86.6, 0] -> raw v_exp_f32 needs no subnormal fixup.
#define K_A      43.280851226668904f

// ---- workspace layout (bytes) ----
constexpr size_t WS_WN = 0;                         // 50000*128 f16 = 12,800,000
constexpr size_t WS_FN = 12800000;                  // 1024*128 f16  =    262,144
constexpr size_t WS_T  = WS_FN + 262144;            // 1024 f32 target logits
constexpr size_t WS_C  = WS_T  + 4096;              // 1024 f32 center-loss partials
constexpr size_t WS_P  = WS_C  + 4096;              // 1024*64 f32 sum-exp partials

__device__ __forceinline__ float wred(float v) {
#pragma unroll
  for (int o = 16; o > 0; o >>= 1) v += __shfl_xor(v, o, 32);
  return v;
}

// Kernel 1: normalize weight rows -> f16.  One wave per row, float4 per lane.
__global__ __launch_bounds__(256) void wnorm_kernel(const float* __restrict__ w,
                                                    _Float16* __restrict__ wn) {
  const int wave = threadIdx.x >> 5, lane = threadIdx.x & 31;
  const int row = blockIdx.x * 8 + wave;                 // 6250*8 = 50000 exact
  const float4 v = ((const float4*)(w + (size_t)row * DIM))[lane];
  float ss = wred(v.x * v.x + v.y * v.y + v.z * v.z + v.w * v.w);
  const float r = 1.0f / fmaxf(sqrtf(ss), 1e-8f);
  v4h o;
  o[0] = (_Float16)(v.x * r); o[1] = (_Float16)(v.y * r);
  o[2] = (_Float16)(v.z * r); o[3] = (_Float16)(v.w * r);
  ((v4h*)(wn + (size_t)row * DIM))[lane] = o;
}

// Kernel 2: normalize features -> f16; exact fp32 target cosine + center loss.
__global__ __launch_bounds__(256) void fprep_kernel(const float* __restrict__ f,
                                                    const float* __restrict__ w,
                                                    const int* __restrict__ label,
                                                    _Float16* __restrict__ fn,
                                                    float* __restrict__ t_row,
                                                    float* __restrict__ c_row) {
  const int wave = threadIdx.x >> 5, lane = threadIdx.x & 31;
  const int row = blockIdx.x * 8 + wave;                 // 128*8 = 1024 exact
  const float4 fv = ((const float4*)(f + (size_t)row * DIM))[lane];
  const int y = label[row];
  const float4 wv = ((const float4*)(w + (size_t)y * DIM))[lane];
  float ssf = fv.x * fv.x + fv.y * fv.y + fv.z * fv.z + fv.w * fv.w;
  float ssw = wv.x * wv.x + wv.y * wv.y + wv.z * wv.z + wv.w * wv.w;
  float dfw = fv.x * wv.x + fv.y * wv.y + fv.z * wv.z + fv.w * wv.w;
  float dx = fv.x - wv.x, dy = fv.y - wv.y, dz = fv.z - wv.z, dw = fv.w - wv.w;
  float ssd = dx * dx + dy * dy + dz * dz + dw * dw;
  ssf = wred(ssf); ssw = wred(ssw); dfw = wred(dfw); ssd = wred(ssd);
  const float rf = 1.0f / fmaxf(sqrtf(ssf), 1e-8f);
  const float rw = 1.0f / fmaxf(sqrtf(ssw), 1e-8f);
  if (lane == 0) {
    t_row[row] = S_SCALE * (dfw * rf * rw - MARGIN);     // target logit (exact)
    c_row[row] = ssd;                                    // ||f - w_y||^2
  }
  v4h o;
  o[0] = (_Float16)(fv.x * rf); o[1] = (_Float16)(fv.y * rf);
  o[2] = (_Float16)(fv.z * rf); o[3] = (_Float16)(fv.w * rf);
  ((v4h*)(fn + (size_t)row * DIM))[lane] = o;
}

// Kernel 3: fused f16-WMMA GEMM + fixed-max sum-of-exp (hard bound 30).
// Block = 8 waves = 256 feature rows; ALL waves walk the SAME column-tile
// stream (identical B addresses -> WGP$ broadcast, 8x less L2 traffic),
// each wave owning a 32-row slice.  Ping-pong register-buffered B.
__global__ __launch_bounds__(256) void gemm_lse_kernel(const _Float16* __restrict__ fn,
                                                       const _Float16* __restrict__ wn,
                                                       float* __restrict__ partials) {
  const int wave = threadIdx.x >> 5, lane = threadIdx.x & 31;
  const int rb = blockIdx.x, by = blockIdx.y;
  const int l15 = lane & 15;
  const int hi = (lane >> 4) * 8;            // A: K half-offset per lane half
  const int boff = (lane < 16) ? 0 : 16;     // B: K 0-15 / 16-31 per lane half

  union V16 { v16h v; v8h h[2]; };
  V16 a0[4], a1[4];                          // A fragments: loop-invariant, in VGPRs
  {
    const _Float16* p0 = fn + (size_t)(rb * 256 + wave * 32 + l15) * DIM;
    const _Float16* p1 = p0 + 16 * DIM;
#pragma unroll
    for (int kb = 0; kb < 4; ++kb) {
      a0[kb].h[0] = *(const v8h*)(p0 + kb * 32 + 2 * hi);
      a0[kb].h[1] = *(const v8h*)(p0 + kb * 32 + 8 + 2 * hi);
      a1[kb].h[0] = *(const v8h*)(p1 + kb * 32 + 2 * hi);
      a1[kb].h[1] = *(const v8h*)(p1 + kb * 32 + 8 + 2 * hi);
    }
  }

  float rs0[8], rs1[8];                      // per-lane sum-of-exp accumulators
#pragma unroll
  for (int r = 0; r < 8; ++r) { rs0[r] = 0.f; rs1[r] = 0.f; }

  auto loadB = [&](v16h* b, const _Float16* wp) {
    b[0] = *(const v16h*)(wp);
    b[1] = *(const v16h*)(wp + 32);
    b[2] = *(const v16h*)(wp + 64);
    b[3] = *(const v16h*)(wp + 96);
  };
  auto compute = [&](const v16h* b) {
    v8f c0 = {}, c1 = {};
#pragma unroll
    for (int kb = 0; kb < 4; ++kb) {
      c0 = __builtin_amdgcn_wmma_f32_16x16x32_f16(false, a0[kb].v, false, b[kb],
                                                  (short)0, c0, false, false);
      c1 = __builtin_amdgcn_wmma_f32_16x16x32_f16(false, a1[kb].v, false, b[kb],
                                                  (short)0, c1, false, false);
    }
#pragma unroll
    for (int r = 0; r < 8; ++r) {            // s += 2^(fma(c, 30*log2e, -30*log2e))
      rs0[r] += __builtin_amdgcn_exp2f(fmaf(c0[r], K_A, -K_A));
      rs1[r] += __builtin_amdgcn_exp2f(fmaf(c1[r], K_A, -K_A));
    }
  };

  int rem = (NTILES - by + CCH - 1) / CCH - 1;           // tiles after first (>=47)
  const size_t STEP = (size_t)CCH * 16 * DIM;            // halves per tile stride
  const _Float16* wp = wn + (size_t)(by * 16 + l15) * DIM + boff;

  v16h b[4], nb[4];
  loadB(b, wp); wp += STEP;
  while (rem >= 2) {                         // ping-pong: 2 tiles / iteration
    loadB(nb, wp);
    compute(b);
    loadB(b, wp + STEP);
    compute(nb);
    wp += 2 * STEP;
    rem -= 2;
  }
  if (rem == 1) { loadB(nb, wp); compute(b); compute(nb); }
  else          { compute(b); }

  // in-wave reduce: sum the 16 column-lanes of each half (xor < 16 stays in half)
#pragma unroll
  for (int r = 0; r < 8; ++r) {
#pragma unroll
    for (int o = 1; o < 16; o <<= 1) {
      rs0[r] += __shfl_xor(rs0[r], o, 32);
      rs1[r] += __shfl_xor(rs1[r], o, 32);
    }
  }
  if (l15 == 0) {                            // lanes 0 and 16 hold row sums
    const int mb = (lane < 16) ? 0 : 8;      // C layout: VGPR r -> M = mb + r
    const int base = rb * 256 + wave * 32 + mb;
#pragma unroll
    for (int r = 0; r < 8; ++r) {
      partials[(size_t)(base + r) * CCH + by]      = rs0[r];
      partials[(size_t)(base + 16 + r) * CCH + by] = rs1[r];
    }
  }
}

// Kernel 4: merge C-chunk partials, apply margin correction, reduce loss.
__global__ __launch_bounds__(1024) void finalize_kernel(const float* __restrict__ partials,
                                                        const float* __restrict__ t_row,
                                                        const float* __restrict__ c_row,
                                                        float* __restrict__ out) {
  __shared__ float sv[1024];
  __shared__ float sc[1024];
  const int i = threadIdx.x;
  float S = 0.f;
  for (int j = 0; j < CCH; ++j) S += partials[(size_t)i * CCH + j];
  const float t  = t_row[i];                 // s*(cos_y - m), exact fp32
  const float xa = t + S_SCALE * MARGIN;     // s*cos_y (unmargined term inside S)
  S = fmaxf(S + __expf(t - S_SCALE) - __expf(xa - S_SCALE), 1e-30f);
  sv[i] = (S_SCALE + __logf(S)) - t;         // lse - target_logit
  sc[i] = c_row[i];
  __syncthreads();
  for (int st = 512; st > 0; st >>= 1) {
    if (i < st) { sv[i] += sv[i + st]; sc[i] += sc[i + st]; }
    __syncthreads();
  }
  if (i == 0) out[0] = sv[0] * (1.0f / (float)NROWS) + LAMBDA * 0.5f * sc[0];
}

extern "C" void kernel_launch(void* const* d_in, const int* in_sizes, int n_in,
                              void* d_out, int out_size, void* d_ws, size_t ws_size,
                              hipStream_t stream) {
  const float* feature = (const float*)d_in[0];
  const float* weight  = (const float*)d_in[1];
  const int*   label   = (const int*)d_in[2];
  float* out = (float*)d_out;
  char* ws = (char*)d_ws;
  _Float16* wn   = (_Float16*)(ws + WS_WN);
  _Float16* fn   = (_Float16*)(ws + WS_FN);
  float*    trow = (float*)(ws + WS_T);
  float*    crow = (float*)(ws + WS_C);
  float*    part = (float*)(ws + WS_P);

  wnorm_kernel<<<NCLS / 8, 256, 0, stream>>>(weight, wn);
  fprep_kernel<<<NROWS / 8, 256, 0, stream>>>(feature, weight, label, fn, trow, crow);
  gemm_lse_kernel<<<dim3(RB, CCH), 256, 0, stream>>>(fn, wn, part);
  finalize_kernel<<<1, 1024, 0, stream>>>(part, trow, crow, out);
}
